// CombinedMLPMoEModel_38912403702049
// MI455X (gfx1250) — compile-verified, matrix operand order checked by reading, not run on hardware
//
#include <hip/hip_runtime.h>

// ---------------------------------------------------------------------------
// CDNA5 (gfx1250): all large GEMMs via v_wmma_f32_16x16x32_bf16.
// - f32 activations converted to bf16 with native cvt while staging into LDS
//   (vectorized b128 loads -> b64 packed stores).
// - bf16 weight tiles staged via GLOBAL_LOAD_ASYNC_TO_LDS_B128 when the
//   toolchain exposes the builtin (ASYNCcnt path), else plain b128 copies.
// - MoE: per-expert f32 accumulators, gate weights applied once in epilogue.
// ---------------------------------------------------------------------------

typedef __attribute__((ext_vector_type(16))) __bf16         v16bf;
typedef __attribute__((ext_vector_type(8)))  float          v8f;
typedef __attribute__((ext_vector_type(4)))  unsigned int   u32x4;
typedef __attribute__((ext_vector_type(4)))  float          f32x4;
typedef __attribute__((ext_vector_type(4)))  unsigned short u16x4;
typedef int v4i __attribute__((vector_size(16)));   // matches async builtin param

__device__ __forceinline__ unsigned short f2bf(float x) {
  return __builtin_bit_cast(unsigned short, (__bf16)x);   // native cvt, RNE
}

#if defined(__gfx1250__) && \
    __has_builtin(__builtin_amdgcn_global_load_async_to_lds_b128) && \
    __has_builtin(__builtin_amdgcn_s_wait_asynccnt)
#define USE_ASYNC_LDS 1
#else
#define USE_ASYNC_LDS 0
#endif

#if USE_ASYNC_LDS
__device__ __forceinline__ void async_cp16(const unsigned short* g, unsigned short* l) {
  __builtin_amdgcn_global_load_async_to_lds_b128(
      (__attribute__((address_space(1))) v4i*)g,
      (__attribute__((address_space(3))) v4i*)l, 0, 0);
}
#endif

union Frag {
  v16bf bf;
  u32x4 q[2];
};

// 16x32 bf16 fragment from an LDS tile with row stride 48 halfwords (96B).
// Lane l (0-15): row l, K {0..7, 16..23}; lane l+16: row l, K {8..15, 24..31}.
__device__ __forceinline__ v16bf load_frag(const unsigned short* lds, int row0) {
  const int lane = threadIdx.x & 31;
  const unsigned short* p = lds + (row0 + (lane & 15)) * 48 + ((lane >> 4) << 3);
  Frag f;
  f.q[0] = *(const u32x4*)p;        // K base + {0..7}
  f.q[1] = *(const u32x4*)(p + 16); // K base + {16..23}
  return f.bf;
}

#define WMMA_BF16(A_, B_, C_) \
  __builtin_amdgcn_wmma_f32_16x16x32_bf16(false, (A_), false, (B_), (short)0, (C_), false, false)

#define BN_SCALE 0.99999500003749973f   // 1/sqrt(1+1e-5)

// ---------------------------------------------------------------------------
// Weight transpose-convert: f32 W[K][N] -> bf16 Wt[N][K]
// ---------------------------------------------------------------------------
__global__ void __launch_bounds__(256) k_tconv(const float* __restrict__ W,
                                               unsigned short* __restrict__ Wt,
                                               int K, int N) {
  int id = blockIdx.x * 256 + threadIdx.x;
  if (id >= K * N) return;
  int k = id / N, n = id - k * N;
  Wt[n * K + k] = f2bf(W[id]);
}

// ---------------------------------------------------------------------------
// Generic WMMA GEMM: C[:, coff:coff+N] = epilogue(A(f32,MxK) @ Bt(bf16,[N][K]))
// Block tile 128x128, K step 32. 8 waves: 4(M) x 2(N), wave tile 32x64.
// EPI: 0 = +bias ; 1 = relu(bn(+bias)) ; 2 = bn(+bias)
// ---------------------------------------------------------------------------
template <int EPI>
__global__ void __launch_bounds__(256) k_gemm(
    const float* __restrict__ A, int lda,
    const unsigned short* __restrict__ Bt,
    const float* __restrict__ bias,
    const float* __restrict__ gam, const float* __restrict__ bet,
    float* __restrict__ C, int ldc, int coff, int K) {
  __shared__ __align__(16) unsigned short As[128 * 48];
  __shared__ __align__(16) unsigned short Bs[128 * 48];
  const int tid = threadIdx.x, lane = tid & 31, wave = tid >> 5;
  const int wm = wave & 3, wn = wave >> 2;
  const int bm = blockIdx.x * 128, bn = blockIdx.y * 128;

  v8f acc[2][4];
  const v8f vz = {0.f, 0.f, 0.f, 0.f, 0.f, 0.f, 0.f, 0.f};
#pragma unroll
  for (int i = 0; i < 2; ++i)
#pragma unroll
    for (int j = 0; j < 4; ++j) acc[i][j] = vz;

  for (int k0 = 0; k0 < K; k0 += 32) {
    // --- B tile: 128 rows x 64B, 512 16-byte chunks ---
#pragma unroll
    for (int i = 0; i < 2; ++i) {
      int c2 = tid + (i << 8);
      int r = c2 >> 2, ci = (c2 & 3) << 3;
#if USE_ASYNC_LDS
      async_cp16(Bt + (bn + r) * K + k0 + ci, Bs + r * 48 + ci);
#else
      *(u32x4*)(Bs + r * 48 + ci) = *(const u32x4*)(Bt + (bn + r) * K + k0 + ci);
#endif
    }
    // --- A tile: 128x32 f32 -> bf16, 1024 float4 chunks ---
#pragma unroll
    for (int i = 0; i < 4; ++i) {
      int c4 = tid + (i << 8);
      int r = c4 >> 3, ci = (c4 & 7) << 2;
      f32x4 v = *(const f32x4*)(A + (bm + r) * lda + k0 + ci);
      u16x4 p = {f2bf(v.x), f2bf(v.y), f2bf(v.z), f2bf(v.w)};
      *(u16x4*)(As + r * 48 + ci) = p;
    }
#if USE_ASYNC_LDS
    __builtin_amdgcn_s_wait_asynccnt(0);
#endif
    __syncthreads();
    v16bf a0 = load_frag(As, wm * 32);
    v16bf a1 = load_frag(As, wm * 32 + 16);
#pragma unroll
    for (int j = 0; j < 4; ++j) {
      v16bf b = load_frag(Bs, wn * 64 + j * 16);
      acc[0][j] = WMMA_BF16(a0, b, acc[0][j]);
      acc[1][j] = WMMA_BF16(a1, b, acc[1][j]);
    }
    __syncthreads();
  }

#pragma unroll
  for (int i = 0; i < 2; ++i)
#pragma unroll
    for (int j = 0; j < 4; ++j)
#pragma unroll
      for (int v = 0; v < 8; ++v) {
        int row = bm + wm * 32 + i * 16 + v + ((lane >> 4) << 3);
        int col = bn + wn * 64 + j * 16 + (lane & 15);
        float x = acc[i][j][v] + bias[col];
        if (EPI == 1) x = fmaxf(fmaf(x * BN_SCALE, gam[col], bet[col]), 0.f);
        if (EPI == 2) x = fmaf(x * BN_SCALE, gam[col], bet[col]);
        C[row * ldc + coff + col] = x;
      }
}

// ---------------------------------------------------------------------------
// LayerNorm over rows of m (B x 1536), in place. One block per row.
// ---------------------------------------------------------------------------
__global__ void __launch_bounds__(256) k_ln(float* __restrict__ m,
                                            const float* __restrict__ g,
                                            const float* __restrict__ b) {
  const int row = blockIdx.x, tid = threadIdx.x;
  float v[6], s = 0.f, ss = 0.f;
#pragma unroll
  for (int i = 0; i < 6; ++i) {
    float x = m[row * 1536 + tid + i * 256];
    v[i] = x; s += x; ss += x * x;
  }
  __shared__ float rs[256], rss[256];
  rs[tid] = s; rss[tid] = ss;
  __syncthreads();
  for (int st = 128; st > 0; st >>= 1) {
    if (tid < st) { rs[tid] += rs[tid + st]; rss[tid] += rss[tid + st]; }
    __syncthreads();
  }
  float mean = rs[0] * (1.f / 1536.f);
  float var  = rss[0] * (1.f / 1536.f) - mean * mean;
  float inv  = rsqrtf(var + 1e-5f);
#pragma unroll
  for (int i = 0; i < 6; ++i) {
    int c = tid + i * 256;
    m[row * 1536 + c] = (v[i] - mean) * inv * g[c] + b[c];
  }
}

// ---------------------------------------------------------------------------
// Gating: one wave per (token, slice). logits = m_slice @ Wg + bg, top-2
// softmax; writes dense gate vector (0 for unselected experts).
// ---------------------------------------------------------------------------
__global__ void __launch_bounds__(256) k_gate(const float* __restrict__ m,
                                              const float* __restrict__ Wg,
                                              const float* __restrict__ bg,
                                              float* __restrict__ gw) {
  const int lane = threadIdx.x & 31;
  const int token = blockIdx.x * 8 + (threadIdx.x >> 5);
  const int slice = blockIdx.y;
  const float* x = m + token * 1536 + slice * 512;
  float lg[8];
#pragma unroll
  for (int e = 0; e < 8; ++e) lg[e] = 0.f;
  for (int k = lane; k < 512; k += 32) {
    float xv = x[k];
#pragma unroll
    for (int e = 0; e < 8; ++e) lg[e] = fmaf(xv, Wg[k * 8 + e], lg[e]);
  }
#pragma unroll
  for (int e = 0; e < 8; ++e)
    for (int off = 16; off > 0; off >>= 1) lg[e] += __shfl_down(lg[e], off, 32);
  if (lane == 0) {
#pragma unroll
    for (int e = 0; e < 8; ++e) lg[e] += bg[e];
    int i1 = 0; float v1 = lg[0];
#pragma unroll
    for (int e = 1; e < 8; ++e) if (lg[e] > v1) { v1 = lg[e]; i1 = e; }
    int i2 = -1; float v2 = -3.4e38f;
#pragma unroll
    for (int e = 0; e < 8; ++e) if (e != i1 && lg[e] > v2) { v2 = lg[e]; i2 = e; }
    float t  = __expf(v2 - v1);
    float w1 = 1.f / (1.f + t), w2 = t / (1.f + t);
    float* o = gw + token * 24 + slice * 8;
#pragma unroll
    for (int e = 0; e < 8; ++e) o[e] = (e == i1) ? w1 : ((e == i2) ? w2 : 0.f);
  }
}

// ---------------------------------------------------------------------------
// MoE: out_slice = sum_e gw[:,e] * (m_slice @ We[e] + bexp[e]).
// Block tile 64(M) x 64(N); all 8 expert B tiles staged per k-step; one f32
// accumulator pair per expert per wave (8x2 v8f = 128 VGPRs); gate weights
// applied once in the epilogue. Inner loop is pure ds_load + v_wmma.
// 8 waves: 4(M) x 2(N), wave tile 16x32 -> 16 wmma per k-step per wave.
// ---------------------------------------------------------------------------
__global__ void __launch_bounds__(256, 1) k_moe(
    const float* __restrict__ m,              // B x 1536
    const float* __restrict__ gw,             // B x 3 x 8
    const unsigned short* __restrict__ WeT,   // 8 x [N=512][K=512] bf16
    const float* __restrict__ bexp,           // 8 x 512
    float* __restrict__ out) {                // B x 1536
  __shared__ __align__(16) unsigned short As[64 * 48];
  __shared__ __align__(16) unsigned short Bs[8 * 64 * 48];
  __shared__ float gws[64 * 8];
  const int tid = threadIdx.x, lane = tid & 31, wave = tid >> 5;
  const int wm = wave & 3, wn = wave >> 2;
  const int bm = blockIdx.x * 64, bn = blockIdx.y * 64, slice = blockIdx.z;

#pragma unroll
  for (int i = 0; i < 2; ++i) {
    int e = tid + (i << 8);
    if (e < 512) gws[e] = gw[(bm + (e >> 3)) * 24 + slice * 8 + (e & 7)];
  }

  const v8f vz = {0.f, 0.f, 0.f, 0.f, 0.f, 0.f, 0.f, 0.f};
  v8f tacc[8][2];
#pragma unroll
  for (int e = 0; e < 8; ++e) { tacc[e][0] = vz; tacc[e][1] = vz; }

  for (int k0 = 0; k0 < 512; k0 += 32) {
    // --- 8 expert B tiles: 2048 16-byte chunks ---
#pragma unroll
    for (int i = 0; i < 8; ++i) {
      int c2 = tid + (i << 8);
      int ex = c2 >> 8, rem = c2 & 255;
      int r = rem >> 2, ci = (rem & 3) << 3;
#if USE_ASYNC_LDS
      async_cp16(WeT + ex * 262144 + (bn + r) * 512 + k0 + ci,
                 Bs + ex * 3072 + r * 48 + ci);
#else
      *(u32x4*)(Bs + ex * 3072 + r * 48 + ci) =
          *(const u32x4*)(WeT + ex * 262144 + (bn + r) * 512 + k0 + ci);
#endif
    }
    // --- A tile: 64x32 f32 -> bf16, 512 float4 chunks ---
#pragma unroll
    for (int i = 0; i < 2; ++i) {
      int c4 = tid + (i << 8);
      int r = c4 >> 3, ci = (c4 & 7) << 2;
      f32x4 v = *(const f32x4*)(m + (bm + r) * 1536 + slice * 512 + k0 + ci);
      u16x4 p = {f2bf(v.x), f2bf(v.y), f2bf(v.z), f2bf(v.w)};
      *(u16x4*)(As + r * 48 + ci) = p;
    }
#if USE_ASYNC_LDS
    __builtin_amdgcn_s_wait_asynccnt(0);
#endif
    __syncthreads();
    v16bf a = load_frag(As, wm * 16);
#pragma unroll
    for (int e = 0; e < 8; ++e) {
      v16bf b0 = load_frag(Bs + e * 3072, wn * 32);
      v16bf b1 = load_frag(Bs + e * 3072, wn * 32 + 16);
      tacc[e][0] = WMMA_BF16(a, b0, tacc[e][0]);
      tacc[e][1] = WMMA_BF16(a, b1, tacc[e][1]);
    }
    __syncthreads();
  }

  // Epilogue: out = sum_e gw[row,e] * (tacc_e + bexp_e)
  const int col0 = bn + wn * 32 + (lane & 15);
  const int col1 = col0 + 16;
  float bx0[8], bx1[8];
#pragma unroll
  for (int e = 0; e < 8; ++e) {
    bx0[e] = bexp[e * 512 + col0];
    bx1[e] = bexp[e * 512 + col1];
  }
#pragma unroll
  for (int v = 0; v < 8; ++v) {
    int rowl = wm * 16 + v + ((lane >> 4) << 3);
    float x0 = 0.f, x1 = 0.f;
#pragma unroll
    for (int e = 0; e < 8; ++e) {
      float w = gws[rowl * 8 + e];
      x0 = fmaf(w, tacc[e][0][v] + bx0[e], x0);
      x1 = fmaf(w, tacc[e][1][v] + bx1[e], x1);
    }
    out[(bm + rowl) * 1536 + slice * 512 + col0] = x0;
    out[(bm + rowl) * 1536 + slice * 512 + col1] = x1;
  }
}

// ---------------------------------------------------------------------------
// Final: out[t] = dot(fused[t,:512], Wr) + br. One wave per token.
// ---------------------------------------------------------------------------
__global__ void __launch_bounds__(256) k_final(const float* __restrict__ fused,
                                               const float* __restrict__ Wr,
                                               const float* __restrict__ br,
                                               float* __restrict__ out) {
  const int lane = threadIdx.x & 31;
  const int token = blockIdx.x * 8 + (threadIdx.x >> 5);
  float s = 0.f;
  for (int k = lane; k < 512; k += 32) s = fmaf(fused[token * 512 + k], Wr[k], s);
  for (int off = 16; off > 0; off >>= 1) s += __shfl_down(s, off, 32);
  if (lane == 0) out[token] = s + br[0];
}

// ---------------------------------------------------------------------------
extern "C" void kernel_launch(void* const* d_in, const int* in_sizes, int n_in,
                              void* d_out, int out_size, void* d_ws, size_t ws_size,
                              hipStream_t stream) {
  const int B = 16384;
  const float* x1  = (const float*)d_in[0];
  const float* x2  = (const float*)d_in[1];
  const float* x3  = (const float*)d_in[2];
  const float* Wp1 = (const float*)d_in[3];
  const float* bp1 = (const float*)d_in[4];
  const float* Wp2 = (const float*)d_in[5];
  const float* bp2 = (const float*)d_in[6];
  const float* Wp3 = (const float*)d_in[7];
  const float* bp3 = (const float*)d_in[8];
  const float* W1  = (const float*)d_in[9];
  const float* b1  = (const float*)d_in[10];
  const float* g1  = (const float*)d_in[11];
  const float* be1 = (const float*)d_in[12];
  const float* W2  = (const float*)d_in[13];
  const float* b2  = (const float*)d_in[14];
  const float* g2  = (const float*)d_in[15];
  const float* be2 = (const float*)d_in[16];
  const float* W3  = (const float*)d_in[17];
  const float* b3  = (const float*)d_in[18];
  const float* lng = (const float*)d_in[19];
  const float* lnb = (const float*)d_in[20];
  const float* Wg  = (const float*)d_in[21];
  const float* bg  = (const float*)d_in[22];
  const float* We  = (const float*)d_in[23];
  const float* bexp= (const float*)d_in[24];
  const float* Wf  = (const float*)d_in[25];
  const float* bff = (const float*)d_in[26];
  const float* bng = (const float*)d_in[27];
  const float* bnb = (const float*)d_in[28];
  const float* Wr  = (const float*)d_in[29];
  const float* br  = (const float*)d_in[30];
  float* out = (float*)d_out;

  char* ws = (char*)d_ws;
  size_t off = 0;
  auto take = [&](size_t bytes) -> void* {
    void* p = ws + off;
    off = (off + bytes + 255) & ~(size_t)255;
    return p;
  };
  float* comb  = (float*)take((size_t)B * 1536 * 4); // reused as moe_out
  float* h1b   = (float*)take((size_t)B * 1024 * 4); // reused as fused
  float* h2b   = (float*)take((size_t)B * 1024 * 4);
  float* mbuf  = (float*)take((size_t)B * 1536 * 4);
  float* gwb   = (float*)take((size_t)B * 24 * 4);
  unsigned short* WpT1 = (unsigned short*)take((size_t)512 * 1024 * 2);
  unsigned short* WpT2 = (unsigned short*)take((size_t)512 * 1024 * 2);
  unsigned short* WpT3 = (unsigned short*)take((size_t)512 * 1024 * 2);
  unsigned short* W1T  = (unsigned short*)take((size_t)1024 * 1536 * 2);
  unsigned short* W2T  = (unsigned short*)take((size_t)1024 * 1024 * 2);
  unsigned short* W3T  = (unsigned short*)take((size_t)1536 * 1024 * 2);
  unsigned short* WeT  = (unsigned short*)take((size_t)8 * 512 * 512 * 2);
  unsigned short* WfT  = (unsigned short*)take((size_t)512 * 1536 * 2);

  const dim3 blk(256);

  // Weight transpose-converts (f32 [K][N] -> bf16 [N][K]).
  k_tconv<<<(1024 * 512) / 256, blk, 0, stream>>>(Wp1, WpT1, 1024, 512);
  k_tconv<<<(1024 * 512) / 256, blk, 0, stream>>>(Wp2, WpT2, 1024, 512);
  k_tconv<<<(1024 * 512) / 256, blk, 0, stream>>>(Wp3, WpT3, 1024, 512);
  k_tconv<<<(1536 * 1024) / 256, blk, 0, stream>>>(W1, W1T, 1536, 1024);
  k_tconv<<<(1024 * 1024) / 256, blk, 0, stream>>>(W2, W2T, 1024, 1024);
  k_tconv<<<(1024 * 1536) / 256, blk, 0, stream>>>(W3, W3T, 1024, 1536);
  for (int e = 0; e < 8; ++e)
    k_tconv<<<(512 * 512) / 256, blk, 0, stream>>>(We + (size_t)e * 262144,
                                                   WeT + (size_t)e * 262144, 512, 512);
  k_tconv<<<(1536 * 512) / 256, blk, 0, stream>>>(Wf, WfT, 1536, 512);

  // Projections into the concat buffer (B x 1536).
  k_gemm<0><<<dim3(B / 128, 512 / 128), blk, 0, stream>>>(x1, 1024, WpT1, bp1, nullptr, nullptr, comb, 1536, 0,    1024);
  k_gemm<0><<<dim3(B / 128, 512 / 128), blk, 0, stream>>>(x2, 1024, WpT2, bp2, nullptr, nullptr, comb, 1536, 512,  1024);
  k_gemm<0><<<dim3(B / 128, 512 / 128), blk, 0, stream>>>(x3, 1024, WpT3, bp3, nullptr, nullptr, comb, 1536, 1024, 1024);

  // Hidden layers with fused BN+ReLU.
  k_gemm<1><<<dim3(B / 128, 1024 / 128), blk, 0, stream>>>(comb, 1536, W1T, b1, g1, be1, h1b, 1024, 0, 1536);
  k_gemm<1><<<dim3(B / 128, 1024 / 128), blk, 0, stream>>>(h1b, 1024, W2T, b2, g2, be2, h2b, 1024, 0, 1024);

  // Pre-LN GEMM, then LayerNorm in place.
  k_gemm<0><<<dim3(B / 128, 1536 / 128), blk, 0, stream>>>(h2b, 1024, W3T, b3, nullptr, nullptr, mbuf, 1536, 0, 1024);
  k_ln<<<B, blk, 0, stream>>>(mbuf, lng, lnb);

  // Gating + MoE (moe output reuses the concat buffer).
  k_gate<<<dim3(B / 8, 3), blk, 0, stream>>>(mbuf, Wg, bg, gwb);
  k_moe<<<dim3(B / 64, 512 / 64, 3), blk, 0, stream>>>(mbuf, gwb, WeT, bexp, comb);

  // Fuse GEMM with BN epilogue, then final reduction.
  k_gemm<2><<<dim3(B / 128, 512 / 128), blk, 0, stream>>>(comb, 1536, WfT, bff, bng, bnb, h1b, 512, 0, 1536);
  k_final<<<B / 8, blk, 0, stream>>>(h1b, Wr, br, out);

  (void)in_sizes; (void)n_in; (void)out_size; (void)ws_size;
}
